// GCNWithSkip_3917010174047
// MI455X (gfx1250) — matrix operand order repaired and checked by reading, not compiled, not run on hardware
//
#include <hip/hip_runtime.h>
#include <hip/hip_bf16.h>

typedef __attribute__((ext_vector_type(16))) __bf16 v16bf;
typedef __attribute__((ext_vector_type(8)))  float  v8f;

#define HDIM 256
#define NNODES 100000

__device__ __forceinline__ __bf16 f2bf(float f) {
    unsigned int u = __builtin_bit_cast(unsigned int, f);
    unsigned int r = (u + 0x7FFFu + ((u >> 16) & 1u)) >> 16;   // round-to-nearest-even
    return __builtin_bit_cast(__bf16, (unsigned short)r);
}

__device__ __forceinline__ float gelu_exact(float x) {
    return 0.5f * x * (1.0f + erff(x * 0.70710678118654752f));
}

// One-time weight conversion: src [K x Ncols] f32 (row-major) -> dst [Nrows x K] bf16
// (transposed, so each WMMA B-fragment is 16 contiguous bf16 = one 32B load).
// Rows n >= Ncols are zero-filled (used to pad W_out's 40 cols to 48).
__global__ __launch_bounds__(256)
void convert_wt_kernel(const float* __restrict__ src, __bf16* __restrict__ dst,
                       int K, int Ncols, int Nrows)
{
    int idx = blockIdx.x * 256 + threadIdx.x;
    if (idx >= Nrows * K) return;
    int n = idx / K;
    int k = idx - n * K;
    float v = (n < Ncols) ? src[(size_t)k * Ncols + n] : 0.0f;
    dst[idx] = f2bf(v);
}

// Fused: out = residual + gelu(layer_norm(A@W + bias; gamma,beta))
// A: [N x K] f32; Wt: [256 x K] bf16 (pre-transposed); out/residual: [N x 256] f32.
// Block = 256 threads = 8 waves; block tile = 32 rows x 256 cols (full rows -> in-block LN).
__global__ __launch_bounds__(256)
void gemm_ln_gelu_kernel(const float* __restrict__ A,
                         const __bf16* __restrict__ Wt,
                         const float* __restrict__ bias,
                         const float* __restrict__ gamma,
                         const float* __restrict__ beta,
                         const float* residual,      // may alias out (same elem per thread)
                         float* out,
                         int K)
{
    __shared__ __bf16 aT[32][36];        // [m][k], padded
    __shared__ float  cT[32][HDIM + 8];  // f32 results, padded
    __shared__ float  redS[32][8];
    __shared__ float  redQ[32][8];
    __shared__ float  rowMean[32], rowRstd[32];

    const int tid     = threadIdx.x;
    const int lane    = tid & 31;
    const int wave    = tid >> 5;        // 0..7
    const int rowHalf = wave >> 2;       // rows 16*rowHalf .. +15
    const int colGrp  = wave & 3;        // cols 64*colGrp .. +63
    const int row0    = blockIdx.x * 32;

    v8f acc[4] = {};

    const int mA    = 16 * rowHalf + (lane & 15);  // A-frag row (tile-local)
    const int khalf = (lane >> 4) * 8;             // A-frag K sub-offset
    const int kbB   = (lane >> 4) * 16;            // B-frag K base

    for (int k0 = 0; k0 < K; k0 += 32) {
        // Stage A tile (32x32 f32 -> bf16), coalesced: thread reads k = tid&31
        #pragma unroll
        for (int i = 0; i < 4; ++i) {
            int m = (tid >> 5) + 8 * i;
            int k = tid & 31;
            aT[m][k] = f2bf(A[(size_t)(row0 + m) * K + (k0 + k)]);
        }
        __syncthreads();

        // A fragment, contiguous K order (same as B):
        // lane<16: K in {0..15}; lane>=16: K in {16..31}; M = lane&15
        v16bf afrag;
        #pragma unroll
        for (int i = 0; i < 16; ++i) {
            int k = 2 * khalf + i;
            afrag[i] = aT[mA][k];
        }
        // 4 B fragments straight from pre-transposed bf16 weights (32B aligned, L2-hit).
        // B 32x16: lanes 0-15 K=0..15, lanes 16-31 K=16..31; N = lane&15.
        #pragma unroll
        for (int j = 0; j < 4; ++j) {
            int n = colGrp * 64 + 16 * j + (lane & 15);
            v16bf bfrag = *reinterpret_cast<const v16bf*>(
                Wt + (size_t)n * K + (k0 + kbB));
            acc[j] = __builtin_amdgcn_wmma_f32_16x16x32_bf16(
                false, afrag, false, bfrag, (short)0, acc[j], false, false);
        }
        __syncthreads();
    }

    // Spill accumulators (+bias) to LDS. C layout: VGPR i, lane l -> M = i + 8*(l>>4), N = l&15
    #pragma unroll
    for (int j = 0; j < 4; ++j) {
        int n = colGrp * 64 + 16 * j + (lane & 15);
        float bn = bias[n];
        #pragma unroll
        for (int i = 0; i < 8; ++i) {
            int m = 16 * rowHalf + 8 * (lane >> 4) + i;
            cT[m][n] = acc[j][i] + bn;
        }
    }
    __syncthreads();

    // Row statistics: 8 threads per row, 32 elems each
    {
        int r = tid >> 3, sub = tid & 7;
        float s = 0.f, q = 0.f;
        #pragma unroll
        for (int i = 0; i < 32; ++i) {
            float v = cT[r][sub * 32 + i];
            s += v; q += v * v;
        }
        redS[r][sub] = s; redQ[r][sub] = q;
    }
    __syncthreads();
    if (tid < 32) {
        float s = 0.f, q = 0.f;
        #pragma unroll
        for (int i = 0; i < 8; ++i) { s += redS[tid][i]; q += redQ[tid][i]; }
        float mean = s * (1.0f / 256.0f);
        float var  = q * (1.0f / 256.0f) - mean * mean;
        rowMean[tid] = mean;
        rowRstd[tid] = rsqrtf(var + 1e-5f);
    }
    __syncthreads();

    // Apply LN + GELU + residual, coalesced: thread = column, loop rows
    {
        float gg = gamma[tid];
        float bb = beta[tid];
        for (int i = 0; i < 32; ++i) {
            float v = cT[i][tid];
            float y = (v - rowMean[i]) * rowRstd[i] * gg + bb;
            y = gelu_exact(y);
            size_t o = (size_t)(row0 + i) * HDIM + tid;
            if (residual) y += residual[o];
            out[o] = y;
        }
    }
}

// h[dst] += w * x[src], edge-parallel; 64 lanes/edge, float4 per lane, f32 atomics.
__global__ __launch_bounds__(256)
void spmm_kernel(const float* __restrict__ x,
                 const int* __restrict__ src,
                 const int* __restrict__ dst,
                 const float* __restrict__ w,
                 float* __restrict__ h, int E)
{
    int e = blockIdx.x * 4 + (threadIdx.x >> 6);
    if (e >= E) return;
    int t = threadIdx.x & 63;
    int s = src[e], d = dst[e];
    float we = w[e];
    float4 v = ((const float4*)(x + (size_t)s * HDIM))[t];
    float* hp = h + (size_t)d * HDIM + t * 4;
    unsafeAtomicAdd(hp + 0, v.x * we);
    unsafeAtomicAdd(hp + 1, v.y * we);
    unsafeAtomicAdd(hp + 2, v.z * we);
    unsafeAtomicAdd(hp + 3, v.w * we);
}

// out[N x 40] = A[N x 256] @ W[256 x 40] + b.
// Wt: [48 x 256] bf16 pre-transposed, rows 40..47 zero. Wave computes 16 rows x 48 cols.
__global__ __launch_bounds__(256)
void out_gemm_kernel(const float* __restrict__ A,
                     const __bf16* __restrict__ Wt,
                     const float* __restrict__ bias,
                     float* __restrict__ out)
{
    const int lane = threadIdx.x & 31;
    const int wave = threadIdx.x >> 5;
    int rt = blockIdx.x * 8 + wave;          // wave-uniform guard -> EXEC all-1s at WMMA
    if (rt >= NNODES / 16) return;
    int row0 = rt * 16;

    v8f acc[3] = {};
    const int m     = lane & 15;
    const int khalf = (lane >> 4) * 8;
    const int kbB   = (lane >> 4) * 16;

    for (int k0 = 0; k0 < HDIM; k0 += 32) {
        v16bf afrag;
        #pragma unroll
        for (int i = 0; i < 16; ++i) {
            int k = k0 + 2 * khalf + i;
            afrag[i] = f2bf(A[(size_t)(row0 + m) * HDIM + k]);
        }
        #pragma unroll
        for (int j = 0; j < 3; ++j) {
            int n = 16 * j + (lane & 15);
            v16bf bfrag = *reinterpret_cast<const v16bf*>(
                Wt + (size_t)n * HDIM + (k0 + kbB));
            acc[j] = __builtin_amdgcn_wmma_f32_16x16x32_bf16(
                false, afrag, false, bfrag, (short)0, acc[j], false, false);
        }
    }
    #pragma unroll
    for (int j = 0; j < 3; ++j) {
        int n = 16 * j + (lane & 15);
        if (n < 40) {
            float bn = bias[n];
            #pragma unroll
            for (int i = 0; i < 8; ++i) {
                int mm = 8 * (lane >> 4) + i;
                out[(size_t)(row0 + mm) * 40 + n] = acc[j][i] + bn;
            }
        }
    }
}

extern "C" void kernel_launch(void* const* d_in, const int* in_sizes, int n_in,
                              void* d_out, int out_size, void* d_ws, size_t ws_size,
                              hipStream_t stream) {
    const float* x_in  = (const float*)d_in[0];
    const int*   esrc  = (const int*)  d_in[1];
    const int*   edst  = (const int*)  d_in[2];
    const float* ew    = (const float*)d_in[3];
    const float* W_in  = (const float*)d_in[4];
    const float* b_in  = (const float*)d_in[5];
    const float* g_in  = (const float*)d_in[6];
    const float* be_in = (const float*)d_in[7];
    const float* W_l   = (const float*)d_in[8];
    const float* b_l   = (const float*)d_in[9];
    const float* g_l   = (const float*)d_in[10];
    const float* be_l  = (const float*)d_in[11];
    const float* W_out = (const float*)d_in[12];
    const float* b_out = (const float*)d_in[13];

    const int N = NNODES;
    const int E = in_sizes[1];
    const int F_IN = in_sizes[0] / N;        // 512
    const int C = 40;

    float* outp = (float*)d_out;
    float* xbuf = outp + (size_t)N * C;      // node features live in d_out's second region

    // Workspace layout: h accumulator, then pre-converted bf16 transposed weights.
    float*  hbuf   = (float*)d_ws;                              // N*256 f32 = 102.4 MB
    __bf16* wt_in  = (__bf16*)((char*)d_ws + (size_t)N * HDIM * sizeof(float));
    __bf16* wt_l   = wt_in + (size_t)HDIM * F_IN;               // 4 x [256 x 256]
    __bf16* wt_out = wt_l + (size_t)4 * HDIM * HDIM;            // [48 x 256], rows>=40 zero

    // ---- one-time weight conversion (bf16, transposed) ----
    {
        int n1 = HDIM * F_IN;                 // 256 x 512
        convert_wt_kernel<<<(n1 + 255) / 256, 256, 0, stream>>>(W_in, wt_in, F_IN, HDIM, HDIM);
        int n2 = HDIM * HDIM;                 // per layer 256 x 256
        for (int l = 0; l < 4; ++l)
            convert_wt_kernel<<<(n2 + 255) / 256, 256, 0, stream>>>(
                W_l + (size_t)l * HDIM * HDIM, wt_l + (size_t)l * HDIM * HDIM, HDIM, HDIM, HDIM);
        int n3 = 48 * HDIM;                   // 48 x 256 (padded from 40 cols)
        convert_wt_kernel<<<(n3 + 255) / 256, 256, 0, stream>>>(W_out, wt_out, HDIM, C, 48);
    }

    // x = gelu(ln(x_in @ W_in + b_in))
    gemm_ln_gelu_kernel<<<N / 32, 256, 0, stream>>>(
        x_in, wt_in, b_in, g_in, be_in, nullptr, xbuf, F_IN);

    for (int l = 0; l < 4; ++l) {
        hipMemsetAsync(hbuf, 0, (size_t)N * HDIM * sizeof(float), stream);
        spmm_kernel<<<(E + 3) / 4, 256, 0, stream>>>(xbuf, esrc, edst, ew, hbuf, E);
        // x = x + gelu(ln(h @ W_l[l] + b_l[l]))   (in-place residual: same elem per thread)
        gemm_ln_gelu_kernel<<<N / 32, 256, 0, stream>>>(
            hbuf, wt_l + (size_t)l * HDIM * HDIM, b_l + l * HDIM,
            g_l + l * HDIM, be_l + l * HDIM, xbuf, xbuf, HDIM);
    }

    // logits = x @ W_out + b_out
    out_gemm_kernel<<<(N / 16 + 7) / 8, 256, 0, stream>>>(xbuf, wt_out, b_out, outp);
}
